// InsiderClassifier_43001212567831
// MI455X (gfx1250) — compile-verified, matrix operand order checked by reading, not run on hardware
//
#include <hip/hip_runtime.h>
#include <hip/hip_fp16.h>

typedef __attribute__((ext_vector_type(16))) _Float16 v16h;
typedef __attribute__((ext_vector_type(8)))  float    v8f;

#define NV 170
#define BATCH 32

// ---- WMMA fragment index helpers (CDNA5 ISA 7.12.2, f16 16x16x32) ----
// A (16xK): M = lane&15, K-offset below.  B (Kx16): N = lane&15, same K-offset.
__device__ __forceinline__ int frag_k(int lane, int h) {
    return ((lane >> 4) & 1) * 8 + ((h >> 3) & 1) * 16 + ((h >> 1) & 3) * 2 + (h & 1);
}

__device__ __forceinline__ v8f wmma16(v16h a, v16h b, v8f c) {
    return __builtin_amdgcn_wmma_f32_16x16x32_f16(false, a, false, b, (short)0, c, false, false);
}

__device__ __forceinline__ v8f zero8() {
    v8f z = {0.f, 0.f, 0.f, 0.f, 0.f, 0.f, 0.f, 0.f};
    return z;
}

// =================== graph construction ===================

// normalized cfc rows: U[(p,b,n), d] , d=12
__global__ void k_cfc(const float* __restrict__ hist, const float* __restrict__ glw,
                      float* __restrict__ U) {
    int r = blockIdx.x * blockDim.x + threadIdx.x;
    if (r >= 2 * BATCH * NV) return;
    int p = r / (BATCH * NV);
    int rem = r - p * (BATCH * NV);
    int b = rem / NV, n = rem - (rem / NV) * NV;
    float v[12];
    float ss = 0.f;
#pragma unroll
    for (int d = 0; d < 12; ++d) {
        float t = hist[(b * 12 + d) * NV + n] * glw[p * 12 + d];
        v[d] = t; ss += t * t;
    }
    float inv = 1.f / fmaxf(sqrtf(ss), 1e-12f);
#pragma unroll
    for (int d = 0; d < 12; ++d) U[(size_t)r * 12 + d] = v[d] * inv;
}

// adj[n,m] = mean_b relu( 0.5 * sum_{p,d} U[p,b,n,d] U[p,b,m,d] )
__global__ void k_adj(const float* __restrict__ U, float* __restrict__ adj) {
    int idx = blockIdx.x * blockDim.x + threadIdx.x;
    if (idx >= NV * NV) return;
    int n = idx / NV, m = idx - (idx / NV) * NV;
    float s = 0.f;
    for (int b = 0; b < BATCH; ++b) {
        float att = 0.f;
#pragma unroll
        for (int p = 0; p < 2; ++p) {
            const float* un = U + ((size_t)(p * BATCH + b) * NV + n) * 12;
            const float* um = U + ((size_t)(p * BATCH + b) * NV + m) * 12;
            float d = 0.f;
#pragma unroll
            for (int k = 0; k < 12; ++k) d += un[k] * um[k];
            att += d;
        }
        s += fmaxf(att * 0.5f, 0.f);
    }
    adj[idx] = s * (1.f / (float)BATCH);
}

__global__ void k_deg(const float* __restrict__ adj, float* __restrict__ dis) {
    int n = threadIdx.x;
    if (n >= 176) return;
    float d = 0.f;
    if (n < NV) {
        for (int m = 0; m < NV; ++m) d += 0.5f * (adj[n * NV + m] + adj[m * NV + n]);
    }
    dis[n] = (d > 0.f) ? rsqrtf(d) : 0.f;
}

__global__ void k_lap(const float* __restrict__ adj, const float* __restrict__ dis,
                      float* __restrict__ lap) {
    int idx = blockIdx.x * blockDim.x + threadIdx.x;
    if (idx >= NV * NV) return;
    int n = idx / NV, m = idx - (idx / NV) * NV;
    float a = 0.5f * (adj[n * NV + m] + adj[m * NV + n]);
    lap[idx] = ((n == m) ? 1.f : 0.f) - dis[n] * a * dis[m];
}

// power iteration for emax, then rescale laplacian into gso
__global__ __launch_bounds__(256) void k_emax_gso(const float* __restrict__ lap,
                                                  float* __restrict__ gso) {
    __shared__ float v[NV];
    __shared__ float red[256];
    int tid = threadIdx.x;
    if (tid < NV) v[tid] = 1.f;
    __syncthreads();
    float em = 0.f;
    for (int it = 0; it < 96; ++it) {
        float w = 0.f;
        if (tid < NV) {
            for (int m = 0; m < NV; ++m) w += lap[tid * NV + m] * v[m];
        }
        red[tid] = (tid < NV) ? w * w : 0.f;
        __syncthreads();
        for (int st = 128; st > 0; st >>= 1) {
            if (tid < st) red[tid] += red[tid + st];
            __syncthreads();
        }
        float nrm = sqrtf(red[0]);
        __syncthreads();
        float inv = (nrm > 1e-30f) ? 1.f / nrm : 0.f;
        if (tid < NV) v[tid] = w * inv;
        em = nrm;
        __syncthreads();
    }
    float scale2 = 2.f / fmaxf(em, 1e-6f);
    bool sub = (em >= 2.f);
    for (int i = tid; i < NV * NV; i += 256) {
        int n = i / NV, m = i - (i / NV) * NV;
        float I = (n == m) ? 1.f : 0.f;
        gso[i] = sub ? (lap[i] - I) : (scale2 * lap[i] - I);
    }
}

// =================== fragment prep kernels ===================

// gso -> A-fragment order f16: [mt(11)][kt(6)][lane(32)][h(16)]
__global__ void k_gso_frag(const float* __restrict__ gso, _Float16* __restrict__ frag) {
    int idx = blockIdx.x * blockDim.x + threadIdx.x;
    if (idx >= 11 * 6 * 512) return;
    int h = idx & 15, lane = (idx >> 4) & 31, tt = idx >> 9;
    int kt = tt % 6, mt = tt / 6;
    int M = mt * 16 + (lane & 15);
    int K = kt * 32 + frag_k(lane, h);
    frag[idx] = (_Float16)((M < NV && K < NV) ? gso[M * NV + K] : 0.f);
}

// cheb weight stack (3,16,16) -> B frags [kt(2)][lane][h], K = k*16+ci (48 padded to 64)
__global__ void k_cheb_wfrag(const float* __restrict__ w, _Float16* __restrict__ frag) {
    int idx = blockIdx.x * blockDim.x + threadIdx.x;
    if (idx >= 2 * 512) return;
    int h = idx & 15, lane = (idx >> 4) & 31, kt = idx >> 9;
    int K = kt * 32 + frag_k(lane, h);
    int co = lane & 15;
    int ks = K >> 4, ci = K & 15;
    frag[idx] = (_Float16)((ks < 3) ? w[(ks * 16 + ci) * 16 + co] : 0.f);
}

// GLU conv weights (2Cout,Cin,KT) -> B frags [kt][nt][lane][h], with residual folded
// into the p-columns at the last tap. foldMode 0: identity pad, 1: a3 1x1 conv.
__global__ void k_tconv_wfrag(const float* __restrict__ w, const float* __restrict__ a3w,
                              _Float16* __restrict__ frag, int Cin, int KT, int Cout,
                              int nkt, int nnt, int foldMode) {
    int idx = blockIdx.x * blockDim.x + threadIdx.x;
    int total = nkt * nnt * 512;
    if (idx >= total) return;
    int h = idx & 15, lane = (idx >> 4) & 31, tt = idx >> 9;
    int nt = tt % nnt, kt = tt / nnt;
    int K = kt * 32 + frag_k(lane, h);
    int c = nt * 16 + (lane & 15);
    int Ktot = KT * Cin, Ntot = 2 * Cout;
    float v = 0.f;
    if (K < Ktot && c < Ntot) {
        int tap = K / Cin, ci = K - tap * Cin;
        v = w[(c * Cin + ci) * KT + tap];
        if (c < Cout && tap == KT - 1) {
            if (foldMode == 0) { if (c < Cin && ci == c) v += 1.f; }
            else               { v += a3w[c * Cin + ci]; }
        }
    }
    frag[idx] = (_Float16)v;
}

// fc1 (128,128) -> B frags [kt(4)][nt(8)][lane][h]
__global__ void k_fc1frag(const float* __restrict__ w, _Float16* __restrict__ frag) {
    int idx = blockIdx.x * blockDim.x + threadIdx.x;
    if (idx >= 4 * 8 * 512) return;
    int h = idx & 15, lane = (idx >> 4) & 31, tt = idx >> 9;
    int nt = tt & 7, kt = tt >> 3;
    int K = kt * 32 + frag_k(lane, h);
    int c = nt * 16 + (lane & 15);
    frag[idx] = (_Float16)w[K * 128 + c];
}

// =================== WMMA GLU temporal conv ===================
// out(b,t,n,c) = (conv_p + bias_p + res) * sigmoid(conv_q + bias_q), res folded in weights.
template <int KT, int CIN, int COUT>
__global__ __launch_bounds__(128) void k_glu(const float* __restrict__ X,
                                             const float* __restrict__ bias,
                                             const float* __restrict__ a3b,
                                             const _Float16* __restrict__ wfrag,
                                             float* __restrict__ Y, int T_in) {
    constexpr int KTOT = KT * CIN;
    constexpr int NKT  = (KTOT + 31) / 32;
    constexpr int NNT  = (2 * COUT) / 16;
    constexpr int HALF = COUT / 16;
    const int T_out = T_in - KT + 1;
    const int M_total = BATCH * T_out * NV;
    const int lane = threadIdx.x & 31;
    const int wave = threadIdx.x >> 5;
    const int Mtile = blockIdx.x * 4 + wave;
    if (Mtile * 16 >= M_total) return;   // wave-uniform exit, EXEC stays full for WMMA
    __builtin_prefetch(wfrag, 0, 3);

    // gather im2col A fragments (f32 -> f16)
    v16h av[NKT];
    const int rowA = Mtile * 16 + (lane & 15);
    const bool rowOK = rowA < M_total;
    int bA = 0, tA = 0, nA = 0;
    if (rowOK) {
        bA = rowA / (T_out * NV);
        int rem = rowA - bA * (T_out * NV);
        tA = rem / NV; nA = rem - tA * NV;
    }
#pragma unroll
    for (int kt = 0; kt < NKT; ++kt) {
        v16h a;
#pragma unroll
        for (int h = 0; h < 16; ++h) {
            int K = kt * 32 + frag_k(lane, h);
            float v = 0.f;
            if (rowOK && K < KTOT) {
                int tap = K / CIN, ci = K - tap * CIN;
                v = X[((size_t)(bA * T_in + tA + tap) * NV + nA) * CIN + ci];
            }
            a[h] = (_Float16)v;
        }
        av[kt] = a;
    }

    const int col = lane & 15;
    const int rowBase = Mtile * 16 + ((lane >> 4) << 3);
#pragma unroll 1
    for (int nt = 0; nt < HALF; ++nt) {
        v8f accP = zero8(), accQ = zero8();
#pragma unroll
        for (int kt = 0; kt < NKT; ++kt) {
            v16h bP = *(const v16h*)(wfrag + ((size_t)(kt * NNT + nt) * 32 + lane) * 16);
            v16h bQ = *(const v16h*)(wfrag + ((size_t)(kt * NNT + nt + HALF) * 32 + lane) * 16);
            accP = wmma16(av[kt], bP, accP);
            accQ = wmma16(av[kt], bQ, accQ);
        }
        const int c = nt * 16 + col;
        const float bp = bias[c] + (a3b ? a3b[c] : 0.f);
        const float bq = bias[COUT + c];
#pragma unroll
        for (int r = 0; r < 8; ++r) {
            int row = rowBase + r;
            if (row < M_total) {
                float p = accP[r] + bp;
                float q = accQ[r] + bq;
                float o = p * (1.f / (1.f + __expf(-q)));
                int b = row / (T_out * NV);
                int rem = row - b * (T_out * NV);
                int t = rem / NV, n = rem - (rem / NV) * NV;
                Y[((size_t)(b * T_out + t) * NV + n) * COUT + c] = o;
            }
        }
    }
}

// =================== WMMA Chebyshev graph conv (one block per (b,t)) ===================
__global__ __launch_bounds__(352) void k_cheb(const float* __restrict__ X,
                                              const _Float16* __restrict__ gsoA,
                                              const _Float16* __restrict__ wstk,
                                              const float* __restrict__ cbias,
                                              float* __restrict__ Y, int T) {
    const int bt = blockIdx.x;
    const int b = bt / T, t = bt - b * T;
    __shared__ float Xs0[176][16];
    __shared__ float Xs1[176][16];
    __shared__ float Xs2[176][16];
    __shared__ alignas(32) _Float16 nodefrag[6 * 512];
    __shared__ alignas(32) _Float16 catfrag[11 * 2 * 512];
    const int tid = threadIdx.x;
    const float* xin = X + (size_t)(b * T + t) * NV * 16;

    // X0 rows 0..169: async copy global -> LDS (CDNA5 ASYNCcnt path), 16B per lane.
    // 170 rows * 64B = 680 b128 chunks.
    for (int ch = tid; ch < NV * 4; ch += 352) {
        unsigned lds = (unsigned)(size_t)(&Xs0[0][0]) + (unsigned)(ch * 16);
        const float* gp = xin + ch * 4;
        asm volatile("global_load_async_to_lds_b128 %0, %1, off"
                     :: "v"(lds), "v"(gp) : "memory");
    }
    // zero pad rows 170..175 with plain LDS stores (disjoint from async region)
    for (int i = tid; i < 6 * 16; i += 352) Xs0[NV + (i >> 4)][i & 15] = 0.f;
    asm volatile("s_wait_asynccnt 0x0" ::: "memory");
    __syncthreads();

    for (int i = tid; i < 6 * 512; i += 352) {
        int h = i & 15, ln = (i >> 4) & 31, kt = i >> 9;
        int K = kt * 32 + frag_k(ln, h);
        nodefrag[i] = (_Float16)((K < 176) ? Xs0[K][ln & 15] : 0.f);
    }
    __syncthreads();

    const int wave = tid >> 5, lane = tid & 31;
    const int col = lane & 15;
    const int rowBase = wave * 16 + ((lane >> 4) << 3);

    // X1 = gso @ X0
    v8f acc = zero8();
#pragma unroll
    for (int kt = 0; kt < 6; ++kt) {
        v16h a  = *(const v16h*)(gsoA + ((size_t)(wave * 6 + kt) * 32 + lane) * 16);
        v16h bf = *(const v16h*)(nodefrag + ((size_t)kt * 32 + lane) * 16);
        acc = wmma16(a, bf, acc);
    }
#pragma unroll
    for (int r = 0; r < 8; ++r) Xs1[rowBase + r][col] = acc[r];
    __syncthreads();
    for (int i = tid; i < 6 * 512; i += 352) {
        int h = i & 15, ln = (i >> 4) & 31, kt = i >> 9;
        int K = kt * 32 + frag_k(ln, h);
        nodefrag[i] = (_Float16)((K < 176) ? Xs1[K][ln & 15] : 0.f);
    }
    __syncthreads();

    // X2 = 2 * gso @ X1 - X0
    acc = zero8();
#pragma unroll
    for (int kt = 0; kt < 6; ++kt) {
        v16h a  = *(const v16h*)(gsoA + ((size_t)(wave * 6 + kt) * 32 + lane) * 16);
        v16h bf = *(const v16h*)(nodefrag + ((size_t)kt * 32 + lane) * 16);
        acc = wmma16(a, bf, acc);
    }
#pragma unroll
    for (int r = 0; r < 8; ++r) {
        int row = rowBase + r;
        Xs2[row][col] = 2.f * acc[r] - Xs0[row][col];
    }
    __syncthreads();

    // [X0|X1|X2] (176x48 pad 64) @ Wstack (48x16)
    for (int i = tid; i < 11 * 1024; i += 352) {
        int h = i & 15, ln = (i >> 4) & 31, tt = i >> 9;
        int kt = tt & 1, mt = tt >> 1;
        int K = kt * 32 + frag_k(ln, h);
        int row = mt * 16 + (ln & 15);
        int ks = K >> 4, ci = K & 15;
        float v = 0.f;
        if (ks == 0) v = Xs0[row][ci];
        else if (ks == 1) v = Xs1[row][ci];
        else if (ks == 2) v = Xs2[row][ci];
        catfrag[i] = (_Float16)v;
    }
    __syncthreads();

    acc = zero8();
#pragma unroll
    for (int kt = 0; kt < 2; ++kt) {
        v16h a  = *(const v16h*)(catfrag + ((size_t)(wave * 2 + kt) * 32 + lane) * 16);
        v16h bf = *(const v16h*)(wstk + ((size_t)kt * 32 + lane) * 16);
        acc = wmma16(a, bf, acc);
    }
    float* yout = Y + (size_t)(b * T + t) * NV * 16;
#pragma unroll
    for (int r = 0; r < 8; ++r) {
        int row = rowBase + r;
        if (row < NV) {
            float v = acc[r] + cbias[col] + Xs0[row][col];
            yout[row * 16 + col] = fmaxf(v, 0.f);
        }
    }
}

// =================== layer norm over (N, C) per (b,t), in place ===================
__global__ __launch_bounds__(256) void k_ln(float* __restrict__ X, const float* __restrict__ g,
                                            const float* __restrict__ bv, int C) {
    const int bt = blockIdx.x;
    float* x = X + (size_t)bt * NV * C;
    const int tot = NV * C;
    __shared__ float red[256];
    const int tid = threadIdx.x;
    float s = 0.f;
    for (int i = tid; i < tot; i += 256) s += x[i];
    red[tid] = s; __syncthreads();
    for (int st = 128; st > 0; st >>= 1) { if (tid < st) red[tid] += red[tid + st]; __syncthreads(); }
    float mu = red[0] / (float)tot;
    __syncthreads();
    float s2 = 0.f;
    for (int i = tid; i < tot; i += 256) { float d = x[i] - mu; s2 += d * d; }
    red[tid] = s2; __syncthreads();
    for (int st = 128; st > 0; st >>= 1) { if (tid < st) red[tid] += red[tid + st]; __syncthreads(); }
    float rinv = rsqrtf(red[0] / (float)tot + 1e-12f);
    __syncthreads();
    for (int i = tid; i < tot; i += 256) {
        int n = i / C, c = i - (i / C) * C;
        x[i] = (x[i] - mu) * rinv * g[n * C + c] + bv[n * C + c];
    }
}

// =================== head: oln LN + fc1 (WMMA) + relu + fc2 ===================
__global__ __launch_bounds__(352) void k_head(const float* __restrict__ X,
                                              const float* __restrict__ g,
                                              const float* __restrict__ bv,
                                              const _Float16* __restrict__ fc1frag,
                                              const float* __restrict__ fc1b,
                                              const float* __restrict__ fc2w,
                                              const float* __restrict__ fc2b,
                                              float* __restrict__ out) {
    const int b = blockIdx.x;
    const float* x = X + (size_t)b * NV * 128;
    __shared__ float red[352];
    __shared__ float rowsum[176];
    __shared__ alignas(32) _Float16 afrag[11 * 4 * 512];
    const int tid = threadIdx.x;
    const int TOT = NV * 128;

    float s = 0.f;
    for (int i = tid; i < TOT; i += 352) s += x[i];
    red[tid] = s; __syncthreads();
    for (int sz = 352; sz > 1;) {
        int half = (sz + 1) >> 1;
        if (tid < sz - half) red[tid] += red[tid + half];
        __syncthreads();
        sz = half;
    }
    float mu = red[0] / (float)TOT;
    __syncthreads();
    float s2 = 0.f;
    for (int i = tid; i < TOT; i += 352) { float d = x[i] - mu; s2 += d * d; }
    red[tid] = s2; __syncthreads();
    for (int sz = 352; sz > 1;) {
        int half = (sz + 1) >> 1;
        if (tid < sz - half) red[tid] += red[tid + half];
        __syncthreads();
        sz = half;
    }
    float rinv = rsqrtf(red[0] / (float)TOT + 1e-12f);

    // stage LN-normalized A fragments
    for (int i = tid; i < 11 * 4 * 512; i += 352) {
        int h = i & 15, ln = (i >> 4) & 31, tt = i >> 9;
        int kt = tt & 3, mt = tt >> 2;
        int row = mt * 16 + (ln & 15);
        int K = kt * 32 + frag_k(ln, h);
        float v = 0.f;
        if (row < NV) {
            float xv = x[row * 128 + K];
            v = (xv - mu) * rinv * g[row * 128 + K] + bv[row * 128 + K];
        }
        afrag[i] = (_Float16)v;
    }
    if (tid < 176) rowsum[tid] = 0.f;
    __syncthreads();

    const int wave = tid >> 5, lane = tid & 31;
    const int col = lane & 15;
    const int rowBase = wave * 16 + ((lane >> 4) << 3);
    float partial[8];
#pragma unroll
    for (int r = 0; r < 8; ++r) partial[r] = 0.f;
#pragma unroll 1
    for (int nt = 0; nt < 8; ++nt) {
        v8f acc = zero8();
#pragma unroll
        for (int kt = 0; kt < 4; ++kt) {
            v16h a  = *(const v16h*)(afrag + ((size_t)(wave * 4 + kt) * 32 + lane) * 16);
            v16h bf = *(const v16h*)(fc1frag + ((size_t)(kt * 8 + nt) * 32 + lane) * 16);
            acc = wmma16(a, bf, acc);
        }
        int c = nt * 16 + col;
        float w2 = fc2w[c], b1 = fc1b[c];
#pragma unroll
        for (int r = 0; r < 8; ++r) {
            float hh = fmaxf(acc[r] + b1, 0.f);
            partial[r] += hh * w2;
        }
    }
#pragma unroll
    for (int r = 0; r < 8; ++r) atomicAdd(&rowsum[rowBase + r], partial[r]);
    __syncthreads();
    if (tid < NV) out[b * NV + tid] = rowsum[tid] + fc2b[0];
}

// =================== launch ===================
extern "C" void kernel_launch(void* const* d_in, const int* in_sizes, int n_in,
                              void* d_out, int out_size, void* d_ws, size_t ws_size,
                              hipStream_t stream) {
    (void)in_sizes; (void)n_in; (void)out_size; (void)ws_size;
    const float* hist = (const float*)d_in[0];
    const float* glw  = (const float*)d_in[1];
    const float* c1w  = (const float*)d_in[2];
    const float* c1b  = (const float*)d_in[3];
    const float* ch1w = (const float*)d_in[4];
    const float* ch1b = (const float*)d_in[5];
    const float* c2w  = (const float*)d_in[6];
    const float* c2b  = (const float*)d_in[7];
    const float* ln2g = (const float*)d_in[8];
    const float* ln2b = (const float*)d_in[9];
    const float* c3w  = (const float*)d_in[10];
    const float* c3b  = (const float*)d_in[11];
    const float* a3w  = (const float*)d_in[12];
    const float* a3b  = (const float*)d_in[13];
    const float* ch2w = (const float*)d_in[14];
    const float* ch2b = (const float*)d_in[15];
    const float* c4w  = (const float*)d_in[16];
    const float* c4b  = (const float*)d_in[17];
    const float* ln4g = (const float*)d_in[18];
    const float* ln4b = (const float*)d_in[19];
    const float* ocw  = (const float*)d_in[20];
    const float* ocb  = (const float*)d_in[21];
    const float* olng = (const float*)d_in[22];
    const float* olnb = (const float*)d_in[23];
    const float* fc1w = (const float*)d_in[24];
    const float* fc1b = (const float*)d_in[25];
    const float* fc2w = (const float*)d_in[26];
    const float* fc2b = (const float*)d_in[27];
    float* out = (float*)d_out;

    float* base = (float*)d_ws;
    size_t o = 0;
    auto alloc = [&](size_t nf) { float* p = base + o; o += (nf + 15) & ~(size_t)15; return p; };
    float* U    = alloc(2 * BATCH * NV * 12);
    float* adj  = alloc(NV * NV);
    float* dis  = alloc(176);
    float* lap  = alloc(NV * NV);
    float* gso  = alloc(NV * NV);
    _Float16* gsoA = (_Float16*)alloc(11 * 6 * 512 / 2);
    _Float16* wch1 = (_Float16*)alloc(512);
    _Float16* wch2 = (_Float16*)alloc(512);
    _Float16* wc1  = (_Float16*)alloc(512);
    _Float16* wc2  = (_Float16*)alloc(4096);
    _Float16* wc3  = (_Float16*)alloc(3072);
    _Float16* wc4  = (_Float16*)alloc(4096);
    _Float16* woc  = (_Float16*)alloc(32768);
    _Float16* wfc1 = (_Float16*)alloc(8192);
    float* bufA = alloc((size_t)BATCH * 8 * NV * 64);
    float* bufB = alloc((size_t)BATCH * 8 * NV * 64);

    // graph construction
    k_cfc<<<(2 * BATCH * NV + 255) / 256, 256, 0, stream>>>(hist, glw, U);
    k_adj<<<(NV * NV + 255) / 256, 256, 0, stream>>>(U, adj);
    k_deg<<<1, 192, 0, stream>>>(adj, dis);
    k_lap<<<(NV * NV + 255) / 256, 256, 0, stream>>>(adj, dis, lap);
    k_emax_gso<<<1, 256, 0, stream>>>(lap, gso);

    // fragment prep
    k_gso_frag<<<(11 * 6 * 512 + 255) / 256, 256, 0, stream>>>(gso, gsoA);
    k_cheb_wfrag<<<4, 256, 0, stream>>>(ch1w, wch1);
    k_cheb_wfrag<<<4, 256, 0, stream>>>(ch2w, wch2);
    k_tconv_wfrag<<<4, 256, 0, stream>>>(c1w, nullptr, wc1, 1, 3, 16, 1, 2, 0);
    k_tconv_wfrag<<<32, 256, 0, stream>>>(c2w, nullptr, wc2, 16, 3, 64, 2, 8, 0);
    k_tconv_wfrag<<<24, 256, 0, stream>>>(c3w, a3w, wc3, 64, 3, 16, 6, 2, 1);
    k_tconv_wfrag<<<32, 256, 0, stream>>>(c4w, nullptr, wc4, 16, 3, 64, 2, 8, 0);
    k_tconv_wfrag<<<256, 256, 0, stream>>>(ocw, nullptr, woc, 64, 4, 128, 8, 16, 0);
    k_fc1frag<<<64, 256, 0, stream>>>(fc1w, wfc1);

    // network
    k_glu<3, 1, 16><<<(3400 + 3) / 4, 128, 0, stream>>>(hist, c1b, nullptr, wc1, bufA, 12);
    k_cheb<<<BATCH * 10, 352, 0, stream>>>(bufA, gsoA, wch1, ch1b, bufB, 10);
    k_glu<3, 16, 64><<<(2720 + 3) / 4, 128, 0, stream>>>(bufB, c2b, nullptr, wc2, bufA, 10);
    k_ln<<<BATCH * 8, 256, 0, stream>>>(bufA, ln2g, ln2b, 64);
    k_glu<3, 64, 16><<<(2040 + 3) / 4, 128, 0, stream>>>(bufA, c3b, a3b, wc3, bufB, 8);
    k_cheb<<<BATCH * 6, 352, 0, stream>>>(bufB, gsoA, wch2, ch2b, bufA, 6);
    k_glu<3, 16, 64><<<(1360 + 3) / 4, 128, 0, stream>>>(bufA, c4b, nullptr, wc4, bufB, 6);
    k_ln<<<BATCH * 4, 256, 0, stream>>>(bufB, ln4g, ln4b, 64);
    k_glu<4, 64, 128><<<(340 + 3) / 4, 128, 0, stream>>>(bufB, ocb, nullptr, woc, bufA, 4);
    k_head<<<BATCH, 352, 0, stream>>>(bufA, olng, olnb, wfc1, fc1b, fc2w, fc2b, out);
}
